// RNN_layer_83399674954079
// MI455X (gfx1250) — matrix operand order repaired and checked by reading, not compiled, not run on hardware
//
#include <hip/hip_runtime.h>

typedef __attribute__((ext_vector_type(16))) _Float16 v16h;
typedef __attribute__((ext_vector_type(8)))  float    v8f;

#define PI_F 3.14159265358979f

// -------------------- helpers --------------------

__device__ __forceinline__ float2 cmul2(float2 a, float2 b) {
    return make_float2(a.x * b.x - a.y * b.y, a.x * b.y + a.y * b.x);
}
__device__ __forceinline__ float2 cadd2(float2 a, float2 b) {
    return make_float2(a.x + b.x, a.y + b.y);
}

// phase exponent for amplitude index m (qubit q <-> bit (3-q)) given 7 angles:
// phi_m = 0.5*( sum_q a_q * s_q  -  sum_i a_{4+i} * s_i * s_{i+1} ),  s_q = 2*bit_q - 1
__device__ __forceinline__ float phase_angle(int m, const float* a) {
    float s0 = (m & 8) ? 1.f : -1.f;
    float s1 = (m & 4) ? 1.f : -1.f;
    float s2 = (m & 2) ? 1.f : -1.f;
    float s3 = (m & 1) ? 1.f : -1.f;
    return 0.5f * (a[0] * s0 + a[1] * s1 + a[2] * s2 + a[3] * s3
                   - a[4] * s0 * s1 - a[5] * s1 * s2 - a[6] * s2 * s3);
}

// in-register 8-point Walsh-Hadamard (bits 0..2 of amplitude index)
__device__ __forceinline__ void wht8(float* v) {
    float t;
    t = v[0]; v[0] = t + v[1]; v[1] = t - v[1];
    t = v[2]; v[2] = t + v[3]; v[3] = t - v[3];
    t = v[4]; v[4] = t + v[5]; v[5] = t - v[5];
    t = v[6]; v[6] = t + v[7]; v[7] = t - v[7];
    t = v[0]; v[0] = t + v[2]; v[2] = t - v[2];
    t = v[1]; v[1] = t + v[3]; v[3] = t - v[3];
    t = v[4]; v[4] = t + v[6]; v[6] = t - v[6];
    t = v[5]; v[5] = t + v[7]; v[7] = t - v[7];
    t = v[0]; v[0] = t + v[4]; v[4] = t - v[4];
    t = v[1]; v[1] = t + v[5]; v[5] = t - v[5];
    t = v[2]; v[2] = t + v[6]; v[6] = t - v[6];
    t = v[3]; v[3] = t + v[7]; v[7] = t - v[7];
}

// -------------------- setup: build ansatz unitaries + WMMA A fragments --------------------
// frag layout: [4 fragments][32 lanes][16 halves], fragments:
//   0: A1 rows [Re(A1) -Im(A1)]   1: A1 rows [Im(A1) Re(A1)]
//   2: A2 rows [Re(A2) -Im(A2)]   3: A2 rows [Im(A2) Re(A2)]
// Element e of lane L maps to (M = L%16, K = e + (L<16?0:16))
// (contiguous k order, matching the B fragments).
__global__ void qrnn_setup_kernel(const float* __restrict__ p1,
                                  const float* __restrict__ p2,
                                  _Float16* __restrict__ frag) {
    __shared__ float2 A[2][16][16];
    if (threadIdx.x == 0) {
        for (int w = 0; w < 2; ++w) {
            const float* p = (w == 0) ? p1 : p2;
            for (int m = 0; m < 16; ++m)
                for (int c = 0; c < 16; ++c)
                    A[w][m][c] = make_float2((m == c) ? 1.f : 0.f, 0.f);
            for (int l = 0; l < 2; ++l) {
                // Rot(phi, theta, omega) on each qubit (left-multiply)
                for (int q = 0; q < 4; ++q) {
                    float phi = p[(l * 4 + q) * 3 + 0];
                    float th  = p[(l * 4 + q) * 3 + 1];
                    float om  = p[(l * 4 + q) * 3 + 2];
                    float ch = cosf(0.5f * th), sh = sinf(0.5f * th);
                    float apo = 0.5f * (phi + om), amo = 0.5f * (phi - om);
                    float2 ga = make_float2( cosf(apo) * ch, -sinf(apo) * ch);
                    float2 gb = make_float2(-cosf(amo) * sh, -sinf(amo) * sh);
                    float2 gd = make_float2( cosf(amo) * sh, -sinf(amo) * sh);
                    float2 ge = make_float2( cosf(apo) * ch,  sinf(apo) * ch);
                    int tb = 1 << (3 - q);
                    for (int c = 0; c < 16; ++c)
                        for (int m = 0; m < 16; ++m)
                            if (!(m & tb)) {
                                float2 u0 = A[w][m][c], u1 = A[w][m | tb][c];
                                A[w][m][c]      = cadd2(cmul2(ga, u0), cmul2(gb, u1));
                                A[w][m | tb][c] = cadd2(cmul2(gd, u0), cmul2(ge, u1));
                            }
                }
                // CNOT ring (left-multiply = row swap where control bit set)
                for (int q = 0; q < 4; ++q) {
                    int cb = 1 << (3 - q);
                    int tb = 1 << (3 - ((q + 1) & 3));
                    for (int c = 0; c < 16; ++c)
                        for (int m = 0; m < 16; ++m)
                            if ((m & cb) && !(m & tb)) {
                                float2 tmp = A[w][m][c];
                                A[w][m][c] = A[w][m | tb][c];
                                A[w][m | tb][c] = tmp;
                            }
                }
            }
        }
    }
    __syncthreads();
    int L = threadIdx.x;
    if (L < 32) {
        int M = L & 15;
        for (int f = 0; f < 4; ++f) {
            int wsel = f >> 1;
            bool reblk = ((f & 1) == 0);
            for (int e = 0; e < 16; ++e) {
                int r = e >> 1, hh = e & 1;
                int K = 2 * r + hh + ((L < 16) ? 0 : 16);
                float v;
                if (K < 16) {
                    float2 u = A[wsel][M][K];
                    v = reblk ? u.x : u.y;
                } else {
                    float2 u = A[wsel][M][K - 16];
                    v = reblk ? -u.y : u.x;
                }
                frag[((size_t)f * 32 + L) * 16 + e] = (_Float16)v;
            }
        }
    }
}

// -------------------- main persistent-state QRNN kernel --------------------
__global__ __launch_bounds__(256) void qrnn_main_kernel(
    const float* __restrict__ inputs,       // [B,S,4]
    const float* __restrict__ initial_t,    // [B,7]
    const _Float16* __restrict__ frag,      // [4][32][16]
    const float* __restrict__ kan_base_w,   // [7,4]
    const float* __restrict__ kan_spline_w, // [7,4,4]
    const float* __restrict__ cls_w1,       // [16,4]
    const float* __restrict__ cls_b1,       // [16]
    const float* __restrict__ cls_w2,       // [1,16]
    const float* __restrict__ cls_b2,       // [1]
    float* __restrict__ out, int B, int S) {

    const int lane  = threadIdx.x & 31;
    const int wv    = threadIdx.x >> 5;
    const int group = blockIdx.x * (blockDim.x >> 5) + wv;
    if (group * 16 >= B) return;                 // wave-uniform (EXEC stays all-ones)
    const int  col  = lane & 15;
    const int  b    = group * 16 + col;
    const bool lo   = lane < 16;
    const int  hsel = lo ? 0 : 8;                // this lane's amplitude-half offset

    // constant WMMA A fragments (augmented-real ansatz unitaries)
    v16h a1r = *reinterpret_cast<const v16h*>(frag + (size_t)(0 * 32 + lane) * 16);
    v16h a1i = *reinterpret_cast<const v16h*>(frag + (size_t)(1 * 32 + lane) * 16);
    v16h a2r = *reinterpret_cast<const v16h*>(frag + (size_t)(2 * 32 + lane) * 16);
    v16h a2i = *reinterpret_cast<const v16h*>(frag + (size_t)(3 * 32 + lane) * 16);

    const float4* xrow = reinterpret_cast<const float4*>(inputs + (size_t)b * S * 4);

    float hid0 = 0.f, hid1 = 0.f, hid2 = 0.f, hid3 = 0.f;

    for (int t = 0; t < S; ++t) {
        // ----- hidden-path angles: initial_t at t=0, else KAN(hidden) -----
        float hang[7];
        if (t == 0) {
            #pragma unroll
            for (int i = 0; i < 7; ++i) hang[i] = initial_t[(size_t)b * 7 + i];
        } else {
            float xs[4] = {hid0, hid1, hid2, hid3};
            float si[4], bs[4][4];
            #pragma unroll
            for (int i = 0; i < 4; ++i) {
                float x = xs[i];
                si[i] = x / (1.f + __expf(-x));      // silu
                float bas[7];
                #pragma unroll
                for (int g = 0; g < 7; ++g) {        // order-0 indicators on grid [-7..7], h=2
                    float gl = -7.f + 2.f * g;
                    bas[g] = (x >= gl && x < gl + 2.f) ? 1.f : 0.f;
                }
                #pragma unroll
                for (int k = 1; k <= 3; ++k) {       // Cox-de-Boor up to order 3
                    float inv = 1.f / (2.f * k);
                    #pragma unroll
                    for (int i2 = 0; i2 < 7; ++i2) {
                        if (i2 < 7 - k) {
                            float gi = -7.f + 2.f * i2;
                            bas[i2] = (x - gi) * inv * bas[i2]
                                    + ((gi + 2.f * (k + 1)) - x) * inv * bas[i2 + 1];
                        }
                    }
                }
                bs[i][0] = bas[0]; bs[i][1] = bas[1]; bs[i][2] = bas[2]; bs[i][3] = bas[3];
            }
            #pragma unroll
            for (int o = 0; o < 7; ++o) {
                float acc = 0.f;
                #pragma unroll
                for (int i = 0; i < 4; ++i) acc += si[i] * kan_base_w[o * 4 + i];
                #pragma unroll
                for (int i = 0; i < 4; ++i)
                    #pragma unroll
                    for (int j = 0; j < 4; ++j)
                        acc += bs[i][j] * kan_spline_w[(o * 4 + i) * 4 + j];
                hang[o] = acc;
            }
        }

        // ----- B fragment #1: v = D(hang) * (H^4|0>) = 0.25 * exp(i*phi) -----
        // lanes 0-15 carry Re (K=e), lanes 16-31 carry Im (K=16+e); sin(x+pi/2)=cos(x)
        float shift = lo ? 1.57079632679f : 0.f;
        v16h bh;
        #pragma unroll
        for (int e = 0; e < 16; ++e)
            bh[e] = (_Float16)(0.25f * __sinf(phase_angle(e, hang) + shift));

        v8f zero = {};
        // v1 = A1 * v   (complex via two real WMMAs on augmented blocks)
        v8f dre = __builtin_amdgcn_wmma_f32_16x16x32_f16(false, a1r, false, bh, (short)0, zero, false, false);
        v8f dim = __builtin_amdgcn_wmma_f32_16x16x32_f16(false, a1i, false, bh, (short)0, zero, false, false);

        // ----- v2 = H^4 * v1 : WHT bits0-2 in regs, bit3 across lane halves -----
        float vr[8], vi[8];
        #pragma unroll
        for (int j = 0; j < 8; ++j) { vr[j] = dre[j]; vi[j] = dim[j]; }
        wht8(vr); wht8(vi);
        #pragma unroll
        for (int j = 0; j < 8; ++j) {
            float o1 = __shfl_xor(vr[j], 16, 32);
            float o2 = __shfl_xor(vi[j], 16, 32);
            vr[j] = lo ? vr[j] + o1 : o1 - vr[j];
            vi[j] = lo ? vi[j] + o2 : o2 - vi[j];
        }

        // ----- input embed + phase multiply: v3 = 0.25 * D(xa) * v2 -----
        float4 x4 = xrow[t];
        float xa[7] = {x4.x, x4.y, x4.z, x4.w,
                       (PI_F - x4.x) * (PI_F - x4.y),
                       (PI_F - x4.y) * (PI_F - x4.z),
                       (PI_F - x4.z) * (PI_F - x4.w)};
        float r3[8], i3[8];
        #pragma unroll
        for (int j = 0; j < 8; ++j) {
            int m = j + hsel;
            float s, c;
            __sincosf(phase_angle(m, xa), &s, &c);
            r3[j] = 0.25f * (c * vr[j] - s * vi[j]);
            i3[j] = 0.25f * (s * vr[j] + c * vi[j]);
        }

        // ----- redistribute to B-fragment layout (Re in low half, Im in high half) -----
        v16h bx;
        #pragma unroll
        for (int j = 0; j < 8; ++j) {
            float send = lo ? i3[j] : r3[j];
            float got  = __shfl_xor(send, 16, 32);
            bx[j]     = (_Float16)(lo ? r3[j] : got);
            bx[j + 8] = (_Float16)(lo ? got : i3[j]);
        }

        // v4 = A2 * v3
        v8f fre = __builtin_amdgcn_wmma_f32_16x16x32_f16(false, a2r, false, bx, (short)0, zero, false, false);
        v8f fim = __builtin_amdgcn_wmma_f32_16x16x32_f16(false, a2i, false, bx, (short)0, zero, false, false);

        // ----- measure <Z_q> -----
        float ev0 = 0.f, ev1 = 0.f, ev2 = 0.f, ev3 = 0.f;
        #pragma unroll
        for (int j = 0; j < 8; ++j) {
            int m = j + hsel;
            float p = fre[j] * fre[j] + fim[j] * fim[j];
            ev0 += (m & 8) ? -p : p;
            ev1 += (m & 4) ? -p : p;
            ev2 += (m & 2) ? -p : p;
            ev3 += (m & 1) ? -p : p;
        }
        ev0 += __shfl_xor(ev0, 16, 32);
        ev1 += __shfl_xor(ev1, 16, 32);
        ev2 += __shfl_xor(ev2, 16, 32);
        ev3 += __shfl_xor(ev3, 16, 32);
        hid0 = ev0; hid1 = ev1; hid2 = ev2; hid3 = ev3;
    }

    // ----- classifier head -----
    if (lo && b < B) {
        float acc = cls_b2[0];
        #pragma unroll
        for (int o = 0; o < 16; ++o) {
            float a = cls_b1[o]
                    + cls_w1[o * 4 + 0] * hid0 + cls_w1[o * 4 + 1] * hid1
                    + cls_w1[o * 4 + 2] * hid2 + cls_w1[o * 4 + 3] * hid3;
            acc += cls_w2[o] * fmaxf(a, 0.f);
        }
        out[b] = acc;
    }
}

// -------------------- launch --------------------
extern "C" void kernel_launch(void* const* d_in, const int* in_sizes, int n_in,
                              void* d_out, int out_size, void* d_ws, size_t ws_size,
                              hipStream_t stream) {
    const float* inputs       = (const float*)d_in[0];
    const float* initial_t    = (const float*)d_in[1];
    const float* p1           = (const float*)d_in[2];
    const float* p2           = (const float*)d_in[3];
    const float* kan_base_w   = (const float*)d_in[4];
    const float* kan_spline_w = (const float*)d_in[5];
    const float* cls_w1       = (const float*)d_in[6];
    const float* cls_b1       = (const float*)d_in[7];
    const float* cls_w2       = (const float*)d_in[8];
    const float* cls_b2       = (const float*)d_in[9];
    float* out = (float*)d_out;

    int B = in_sizes[1] / 7;                 // [B,7]
    int S = in_sizes[0] / (B * 4);           // [B,S,4]

    _Float16* frag = (_Float16*)d_ws;        // 4*32*16 halves = 4KB

    qrnn_setup_kernel<<<1, 32, 0, stream>>>(p1, p2, frag);

    int groups = (B + 15) / 16;              // one wave per 16 batch columns
    int waves_per_block = 8;                 // 256 threads
    int blocks = (groups + waves_per_block - 1) / waves_per_block;
    qrnn_main_kernel<<<blocks, waves_per_block * 32, 0, stream>>>(
        inputs, initial_t, frag, kan_base_w, kan_spline_w,
        cls_w1, cls_b1, cls_w2, cls_b2, out, B, S);
}
